// MultiheadSelfAttention_82394652606643
// MI455X (gfx1250) — matrix operand order repaired and checked by reading, not compiled, not run on hardware
//
#include <hip/hip_runtime.h>
#include <hip/hip_bf16.h>

// ---------------------------------------------------------------------------
// Multi-head causal self-attention, N=8 T=1024 D=768 H=12 (dh=64), fp32 I/O.
// Compute path: f16 WMMA (v_wmma_f32_16x16x32_f16) with fp32 accumulation.
//
// Sizing note: all f16 intermediates (~65 MB) fit in the 192 MB L2, so
// separate projection passes re-reading xb are L2-resident; we prefer the
// low-register-pressure 32x64-per-wave GEMM (8 accumulators, no spills) over
// a fused QKV kernel (12+ accumulators) that the allocator spills to scratch.
// ---------------------------------------------------------------------------

#define N_B   8
#define T_S   1024
#define D_M   768
#define H_N   12
#define DH    64
#define NT    (N_B * T_S)          // 8192 rows
#define EB    (NT * D_M)           // elements of one [NT, D] buffer
#define WE    (D_M * D_M)          // elements of one weight matrix

typedef _Float16 f16;
typedef __attribute__((ext_vector_type(16))) _Float16 v16h;
typedef __attribute__((ext_vector_type(8)))  _Float16 v8h;
typedef __attribute__((ext_vector_type(4)))  _Float16 v4h;
typedef __attribute__((ext_vector_type(8)))  float    v8f;

union V16U { v16h v; v8h h[2]; };

// D = A(16x32) x B(32x16) + C, f16 inputs, f32 accumulate
__device__ __forceinline__ v8f wmma16(v16h a, v16h b, v8f c) {
    return __builtin_amdgcn_wmma_f32_16x16x32_f16(
        /*neg_a=*/false, a, /*neg_b=*/false, b,
        /*c_mod=*/(short)0, c, /*reuse_a=*/false, /*reuse_b=*/false);
}

// A-fragment (16x32, f16), source row-major with leading dim `ld`.
// ISA layout: lane L -> row M = L%16; VGPR0..3 hold K = half*8 .. half*8+7,
// VGPR4..7 hold K = 16+half*8 .. 16+half*8+7   (half = L/16).
__device__ __forceinline__ v16h load_a_frag(const f16* __restrict__ tile, int ld) {
    const int lane = threadIdx.x & 31;
    const int m  = lane & 15;
    const int hl = lane >> 4;
    const f16* p = tile + m * ld + hl * 8;
    V16U u;
    u.h[0] = *(const v8h*)(p);
    u.h[1] = *(const v8h*)(p + 16);
    return u.v;
}

// Same, but for LDS (no restrict: the staging buffer is also written through
// another pointer by other lanes; ordering is enforced with explicit barriers).
__device__ __forceinline__ v16h load_a_frag_lds(const f16* tile, int ld) {
    const int lane = threadIdx.x & 31;
    const int m  = lane & 15;
    const int hl = lane >> 4;
    const f16* p = tile + m * ld + hl * 8;
    V16U u;
    u.h[0] = *(const v8h*)(p);
    u.h[1] = *(const v8h*)(p + 16);
    return u.v;
}

// B-fragment (32x16) where B(k,n) = SRC[n][k], SRC row-major leading dim `ld`.
// ISA layout: lane L -> col N = L%16; K = half*8 .. half*8+7, then 16+half*8 .. 16+half*8+7.
__device__ __forceinline__ v16h load_bT_frag(const f16* __restrict__ tile, int ld) {
    const int lane = threadIdx.x & 31;
    const f16* p = tile + (lane & 15) * ld + (lane >> 4) * 8;
    V16U u;
    u.h[0] = *(const v8h*)(p);
    u.h[1] = *(const v8h*)(p + 16);
    return u.v;
}

// ---------------------------------------------------------------------------
// Kernel 0: fp32 -> f16 conversion (vectorized x4). n4 = elements/4.
// ---------------------------------------------------------------------------
__global__ void cvt_f32_to_f16(const float* __restrict__ src, f16* __restrict__ dst, int n4) {
    int i = blockIdx.x * blockDim.x + threadIdx.x;
    if (i < n4) {
        float4 f = ((const float4*)src)[i];
        v4h h = { (f16)f.x, (f16)f.y, (f16)f.z, (f16)f.w };
        ((v4h*)dst)[i] = h;
    }
}

// ---------------------------------------------------------------------------
// Kernel 1: projection  C[m,n] = sum_k A[m,k] * W[n,k]   (y = x @ W^T)
// Per-wave 32x64 tile (2x4 accumulators = 64 VGPRs; no spills).
// mode 0: store f16 row-major [NT,D].
// mode 1: store transposed: Vt[n][d][t] (flat n*D*T + d*T + t), so the PV
//         GEMM's B-fragments are contiguous; each lane's 8 C-elements are
//         contiguous t's -> one packed b128 store per accumulator.
// Block = 128 threads (4 waves stacked along M). Grid = (NT/128, D/64).
// ---------------------------------------------------------------------------
__global__ void proj_gemm(const f16* __restrict__ A, const f16* __restrict__ W,
                          f16* __restrict__ dst, int mode) {
    const int wave = threadIdx.x >> 5;
    const int lane = threadIdx.x & 31;
    const int m    = lane & 15;
    const int hl   = lane >> 4;
    const int m0   = blockIdx.x * 128 + wave * 32;
    const int n0   = blockIdx.y * 64;

    v8f acc[2][4];
#pragma unroll
    for (int i = 0; i < 2; ++i)
#pragma unroll
        for (int c = 0; c < 4; ++c)
            acc[i][c] = v8f{0.f,0.f,0.f,0.f,0.f,0.f,0.f,0.f};

    for (int k = 0; k < D_M; k += 32) {
        v16h af0 = load_a_frag(A + (size_t)(m0     ) * D_M + k, D_M);
        v16h af1 = load_a_frag(A + (size_t)(m0 + 16) * D_M + k, D_M);
#pragma unroll
        for (int c = 0; c < 4; ++c) {
            v16h bf = load_bT_frag(W + (size_t)(n0 + c * 16) * D_M + k, D_M);
            acc[0][c] = wmma16(af0, bf, acc[0][c]);
            acc[1][c] = wmma16(af1, bf, acc[1][c]);
        }
    }

    if (mode == 0) {
        // row-major f16 [NT, D]
#pragma unroll
        for (int i = 0; i < 2; ++i)
#pragma unroll
            for (int c = 0; c < 4; ++c)
#pragma unroll
                for (int r = 0; r < 8; ++r) {
                    int row = m0 + i * 16 + r + 8 * hl;
                    int col = n0 + c * 16 + m;
                    dst[(size_t)row * D_M + col] = (f16)acc[i][c][r];
                }
    } else {
        // transposed: Vt[nb][d][t]; rows r=0..7 are contiguous t -> b128 store
        const int nb = m0 >> 10;           // tile never crosses batch (1024%32==0)
        const int tb = m0 & 1023;
#pragma unroll
        for (int i = 0; i < 2; ++i)
#pragma unroll
            for (int c = 0; c < 4; ++c) {
                v8h pk;
#pragma unroll
                for (int r = 0; r < 8; ++r) pk[r] = (f16)acc[i][c][r];
                int col = n0 + c * 16 + m;
                size_t off = (size_t)nb * D_M * T_S + (size_t)col * T_S
                           + tb + i * 16 + 8 * hl;
                *(v8h*)(dst + off) = pk;
            }
    }
}

// ---------------------------------------------------------------------------
// Kernel 2: fused causal flash attention.
// One wave per 16-query tile of one (n,h). Key loop step 32:
//   S(16x32) via 4 WMMAs over d (K=32 twice), online softmax (shfl_xor row
//   reductions inside 16-lane groups matching the C-fragment layout), P
//   restaged via per-wave LDS into the A-fragment layout, 4 PV WMMAs
//   accumulate O(16x64) in fp32.
// LDS hazard handling: s_wait_dscnt 0 (+compiler memory clobber) between the
// staging stores and fragment loads; a second compiler barrier at the end of
// the loop body prevents next-iteration stores (data-independent of this
// iteration's PV WMMAs) from being hoisted above this iteration's loads --
// per-lane alias analysis cannot see the cross-lane LDS communication.
// Block = 128 threads (4 independent waves). Grid = N*H*(T/16)/4 = 1536.
// ---------------------------------------------------------------------------
__global__ void attn_fused(const f16* __restrict__ Qb, const f16* __restrict__ Kb,
                           const f16* __restrict__ Vt, f16* __restrict__ Ob) {
    __shared__ __align__(32) f16 plds[4][16 * 32];

    const int wave = threadIdx.x >> 5;
    const int lane = threadIdx.x & 31;
    const int m    = lane & 15;
    const int hl   = lane >> 4;

    const int g  = blockIdx.x * 4 + wave;       // 0 .. 6143
    const int nb = g / (H_N * (T_S / 16));
    const int hq = g % (H_N * (T_S / 16));
    const int h  = hq >> 6;                     // 64 q-tiles per head
    const int q0 = (hq & 63) * 16;

    const f16* Qbase = Qb + (size_t)(nb * T_S + q0) * D_M + h * DH;
    // hoist the two Q A-fragments (d = 0..31, 32..63)
    v16h qf0 = load_a_frag(Qbase,      D_M);
    v16h qf1 = load_a_frag(Qbase + 32, D_M);

    float rmax[8], rsum[8];
    v8f o[4];
#pragma unroll
    for (int r = 0; r < 8; ++r) { rmax[r] = -__builtin_inff(); rsum[r] = 0.f; }
#pragma unroll
    for (int c = 0; c < 4; ++c) o[c] = v8f{0.f,0.f,0.f,0.f,0.f,0.f,0.f,0.f};

    f16* my = &plds[wave][0];
    const int kt_max = (q0 + 15) >> 5;          // inclusive causal bound

    for (int kt = 0; kt <= kt_max; ++kt) {
        const int k0 = kt * 32;
        const f16* Kbase = Kb + (size_t)(nb * T_S + k0) * D_M + h * DH;

        v8f s0 = v8f{0.f,0.f,0.f,0.f,0.f,0.f,0.f,0.f};
        v8f s1 = s0;
        {   // keys k0..k0+15 and k0+16..k0+31, contraction over d (2 x K=32)
            v16h b00 = load_bT_frag(Kbase,                 D_M);
            v16h b10 = load_bT_frag(Kbase + 16 * D_M,      D_M);
            v16h b01 = load_bT_frag(Kbase + 32,            D_M);
            v16h b11 = load_bT_frag(Kbase + 16 * D_M + 32, D_M);
            s0 = wmma16(qf0, b00, s0);  s1 = wmma16(qf0, b10, s1);
            s0 = wmma16(qf1, b01, s0);  s1 = wmma16(qf1, b11, s1);
        }

        // scale 1/sqrt(64), causal mask, new running row max
        float nm[8];
#pragma unroll
        for (int r = 0; r < 8; ++r) {
            int qrow = q0 + r + 8 * hl;
            float a = s0[r] * 0.125f;
            float b = s1[r] * 0.125f;
            if (k0 + m      > qrow) a = -__builtin_inff();
            if (k0 + 16 + m > qrow) b = -__builtin_inff();
            s0[r] = a; s1[r] = b;
            float mx = fmaxf(a, b);
#pragma unroll
            for (int off = 1; off < 16; off <<= 1)
                mx = fmaxf(mx, __shfl_xor(mx, off, 32));
            nm[r] = fmaxf(rmax[r], mx);
        }

        // exponentiate, rescale running state, stage P into LDS
#pragma unroll
        for (int r = 0; r < 8; ++r) {
            float scl = (rmax[r] == -__builtin_inff()) ? 0.f : __expf(rmax[r] - nm[r]);
            rmax[r] = nm[r];
            float e0 = (s0[r] == -__builtin_inff()) ? 0.f : __expf(s0[r] - nm[r]);
            float e1 = (s1[r] == -__builtin_inff()) ? 0.f : __expf(s1[r] - nm[r]);
            my[(r + 8 * hl) * 32 + m]      = (f16)e0;
            my[(r + 8 * hl) * 32 + 16 + m] = (f16)e1;
            float ps = e0 + e1;
#pragma unroll
            for (int off = 1; off < 16; off <<= 1)
                ps += __shfl_xor(ps, off, 32);
            rsum[r] = rsum[r] * scl + ps;
#pragma unroll
            for (int c = 0; c < 4; ++c) o[c][r] *= scl;
        }

        // make staged P visible to the fragment loads (hardware + compiler)
        asm volatile("s_wait_dscnt 0" ::: "memory");

        v16h pf = load_a_frag_lds(my, 32);            // P as 16x32 A-fragment
        const f16* Vbase = Vt + (size_t)nb * D_M * T_S + (size_t)(h * DH) * T_S + k0;
#pragma unroll
        for (int c = 0; c < 4; ++c) {
            v16h vf = load_bT_frag(Vbase + (size_t)(c * 16) * T_S, T_S);
            o[c] = wmma16(pf, vf, o[c]);
        }

        // WAR fence: keep next iteration's LDS stores below this iteration's
        // LDS loads in the issued instruction stream.
        asm volatile("" ::: "memory");
    }

    // normalize and store O tile as f16 [NT, D]
#pragma unroll
    for (int c = 0; c < 4; ++c)
#pragma unroll
        for (int r = 0; r < 8; ++r) {
            float v = o[c][r] / rsum[r];
            int row = nb * T_S + q0 + r + 8 * hl;
            int col = h * DH + c * 16 + m;
            Ob[(size_t)row * D_M + col] = (f16)v;
        }
}

// ---------------------------------------------------------------------------
// Kernel 3: output projection  out[m,n] = sum_k O[m,k]*Wo[n,k] + bo[n], fp32.
// Per-wave 32x64 tile, block = 128 threads, grid (NT/128, D/64).
// ---------------------------------------------------------------------------
__global__ void out_proj(const f16* __restrict__ A, const f16* __restrict__ W,
                         const float* __restrict__ bias, float* __restrict__ out) {
    const int wave = threadIdx.x >> 5;
    const int lane = threadIdx.x & 31;
    const int m    = lane & 15;
    const int hl   = lane >> 4;
    const int m0   = blockIdx.x * 128 + wave * 32;
    const int n0   = blockIdx.y * 64;

    v8f acc[2][4];
#pragma unroll
    for (int i = 0; i < 2; ++i)
#pragma unroll
        for (int c = 0; c < 4; ++c)
            acc[i][c] = v8f{0.f,0.f,0.f,0.f,0.f,0.f,0.f,0.f};

    for (int k = 0; k < D_M; k += 32) {
        v16h af0 = load_a_frag(A + (size_t)(m0     ) * D_M + k, D_M);
        v16h af1 = load_a_frag(A + (size_t)(m0 + 16) * D_M + k, D_M);
#pragma unroll
        for (int c = 0; c < 4; ++c) {
            v16h bf = load_bT_frag(W + (size_t)(n0 + c * 16) * D_M + k, D_M);
            acc[0][c] = wmma16(af0, bf, acc[0][c]);
            acc[1][c] = wmma16(af1, bf, acc[1][c]);
        }
    }

#pragma unroll
    for (int i = 0; i < 2; ++i)
#pragma unroll
        for (int c = 0; c < 4; ++c) {
            int col = n0 + c * 16 + m;
            float b = bias[col];
#pragma unroll
            for (int r = 0; r < 8; ++r) {
                int row = m0 + i * 16 + r + 8 * hl;
                out[(size_t)row * D_M + col] = acc[i][c][r] + b;
            }
        }
}

// ---------------------------------------------------------------------------
extern "C" void kernel_launch(void* const* d_in, const int* in_sizes, int n_in,
                              void* d_out, int out_size, void* d_ws, size_t ws_size,
                              hipStream_t stream) {
    const float* x  = (const float*)d_in[0];
    const float* Wq = (const float*)d_in[1];
    const float* Wk = (const float*)d_in[2];
    const float* Wv = (const float*)d_in[3];
    const float* Wo = (const float*)d_in[4];
    const float* bo = (const float*)d_in[5];
    float* out = (float*)d_out;

    // workspace layout (f16): xb | Qb | Kb | Vt | Ob | Wqh | Wkh | Wvh | Woh
    f16* xb  = (f16*)d_ws;
    f16* Qb  = xb  + EB;
    f16* Kb  = Qb  + EB;
    f16* Vt  = Kb  + EB;      // transposed: [n][d][t]
    f16* Ob  = Vt  + EB;
    f16* Wqh = Ob  + EB;
    f16* Wkh = Wqh + WE;
    f16* Wvh = Wkh + WE;
    f16* Woh = Wvh + WE;

    // fp32 -> f16 conversions
    cvt_f32_to_f16<<<(EB / 4 + 255) / 256, 256, 0, stream>>>(x,  xb,  EB / 4);
    cvt_f32_to_f16<<<(WE / 4 + 255) / 256, 256, 0, stream>>>(Wq, Wqh, WE / 4);
    cvt_f32_to_f16<<<(WE / 4 + 255) / 256, 256, 0, stream>>>(Wk, Wkh, WE / 4);
    cvt_f32_to_f16<<<(WE / 4 + 255) / 256, 256, 0, stream>>>(Wv, Wvh, WE / 4);
    cvt_f32_to_f16<<<(WE / 4 + 255) / 256, 256, 0, stream>>>(Wo, Woh, WE / 4);

    // Q/K/V projections (V stored transposed for the PV GEMM)
    dim3 pg(NT / 128, D_M / 64);
    proj_gemm<<<pg, 128, 0, stream>>>(xb, Wqh, Qb, 0);
    proj_gemm<<<pg, 128, 0, stream>>>(xb, Wkh, Kb, 0);
    proj_gemm<<<pg, 128, 0, stream>>>(xb, Wvh, Vt, 1);

    // fused causal attention: N*H*(T/16) = 6144 query tiles, 4 per block
    attn_fused<<<(N_B * H_N * (T_S / 16)) / 4, 128, 0, stream>>>(Qb, Kb, Vt, Ob);

    // final projection + bias, fp32 output
    out_proj<<<dim3(NT / 128, D_M / 64), 128, 0, stream>>>(Ob, Woh, bo, out);
}
